// UnifiedSwinBlock_455266534010
// MI455X (gfx1250) — matrix operand order repaired and checked by reading, not compiled, not run on hardware
//
#include <hip/hip_runtime.h>
#include <hip/hip_bf16.h>

typedef __attribute__((ext_vector_type(16))) _Float16 v16h;
typedef __attribute__((ext_vector_type(8)))  float    v8f;

#define CB    64      // batch
#define CC    192     // channels
#define CH    56      // height
#define CWD   56      // width
#define CHW   (CH*CWD)
#define CNH   6       // heads
#define CHD   32      // head dim
#define CWS   7       // window size
#define CNT   49      // tokens per window
#define CNP   64      // padded tokens per window
#define CMLP  384     // mlp hidden
#define CSH   3       // shift
#define CNW   64      // windows per image
#define CBW   (CB*CNW)
#define CTHR  256

// ---------------- WMMA fragment loaders (CDNA5 16x16x32 f16 layouts) -------

// A fragment: 16x32 f16 tile, row-major source, leading dim ld (in halves).
// lanes 0-15: row M=lane, halves 0..15 -> K=0..15
// lanes 16-31: row M=lane-16, halves 0..15 -> K=16..31
__device__ inline v16h load_a_f16(const _Float16* src, int ld) {
  int lane = threadIdx.x & 31;
  int r  = lane & 15;
  int kb = (lane >> 4) * 16;
  const _Float16* p = src + r * ld + kb;
  v16h a;
#pragma unroll
  for (int i = 0; i < 8; ++i) a[i] = p[i];
#pragma unroll
  for (int i = 0; i < 8; ++i) a[8 + i] = p[8 + i];
  return a;
}

// B fragment where the source is stored transposed: B[k][n] = src[n*ld + k].
// Per-lane data is 16 contiguous halves -> vectorizes to 2x b128 loads.
__device__ inline v16h load_bT_f16(const _Float16* src, int ld) {
  int lane = threadIdx.x & 31;
  int n  = lane & 15;
  int kb = (lane >> 4) * 16;
  const _Float16* p = src + n * ld + kb;
  v16h b;
#pragma unroll
  for (int i = 0; i < 16; ++i) b[i] = p[i];
  return b;
}

// ---------------- Kernel 0: weight swizzle (f32 row-major -> fragment f16) --
// Output layout: fragment f = (kt * (N/16) + nt); halves stored as
// out[(f*32 + lane)*16 + i] = W[(kt*32 + (lane>>4)*16 + i)*N + nt*16 + (lane&15)]
__global__ __launch_bounds__(CTHR) void swizzle_w_kernel(
    const float* __restrict__ W, _Float16* __restrict__ out, int K, int N)
{
  int o = blockIdx.x * CTHR + threadIdx.x;
  if (o >= K * N) return;
  int i    = o & 15;
  int lane = (o >> 4) & 31;
  int tile = o >> 9;
  int ntiles = N >> 4;
  int nt = tile % ntiles;
  int kt = tile / ntiles;
  int n  = lane & 15;
  int kb = (lane >> 4) * 16;
  int k  = kt * 32 + kb + i;
  out[o] = (_Float16)W[(size_t)k * N + nt * 16 + n];
}

// ------------------------- Kernel 1: fused dual LN + shift + partition -----
__global__ __launch_bounds__(CTHR) void swin_ln_kernel(
    const float* __restrict__ query,
    const float* __restrict__ gq,  const float* __restrict__ betaq,
    const float* __restrict__ gkv, const float* __restrict__ betakv,
    _Float16* __restrict__ qn, _Float16* __restrict__ kvn)
{
  __shared__ float xs[CC * CWD];
  __shared__ float mean_s[CWD];
  __shared__ float rstd_s[CWD];

  int bimg = blockIdx.x / CH;
  int hrow = blockIdx.x % CH;
  int tid  = threadIdx.x;

  const float* src = query + (size_t)bimg * CC * CHW + (size_t)hrow * CWD;
  for (int i = tid; i < CC * CWD; i += CTHR) {
    int c = i / CWD, w = i % CWD;
    xs[i] = src[(size_t)c * CHW + w];                 // coalesced along w
  }
  __syncthreads();

  if (tid < CWD) {
    float s = 0.f;
    for (int c = 0; c < CC; ++c) s += xs[c * CWD + tid];
    float m = s * (1.0f / CC);
    float v = 0.f;
    for (int c = 0; c < CC; ++c) { float d = xs[c * CWD + tid] - m; v += d * d; }
    mean_s[tid] = m;
    rstd_s[tid] = rsqrtf(v * (1.0f / CC) + 1e-5f);
  }
  __syncthreads();

  for (int i = tid; i < CC * CWD; i += CTHR) {
    int w = i / CC, c = i % CC;
    float xhat = (xs[c * CWD + w] - mean_s[w]) * rstd_s[w];
    // cyclic shift by -SS folded into destination indexing
    int ps = hrow - CSH; if (ps < 0) ps += CH;
    int qs = w    - CSH; if (qs < 0) qs += CWD;
    int win = bimg * CNW + (ps / CWS) * 8 + (qs / CWS);
    int row = (ps % CWS) * CWS + (qs % CWS);
    size_t o = ((size_t)win * CNT + row) * CC + c;
    qn[o]  = (_Float16)(xhat * gq[c]  + betaq[c]);
    kvn[o] = (_Float16)(xhat * gkv[c] + betakv[c]);
  }
}

// ------------------------- Kernel 2: fused per-window attention ------------
__global__ __launch_bounds__(CTHR) void swin_attn_kernel(
    const _Float16* __restrict__ qn, const _Float16* __restrict__ kvn,
    const v16h* __restrict__ WqF, const float* __restrict__ bq,
    const v16h* __restrict__ WkF, const float* __restrict__ bk,
    const v16h* __restrict__ WvF, const float* __restrict__ bv,
    const float* __restrict__ rpb, const int* __restrict__ rel_idx,
    const float* __restrict__ attn_mask,
    const v16h* __restrict__ WprojF, const float* __restrict__ bproj,
    const float* __restrict__ query, float* __restrict__ x1)
{
  extern __shared__ char smem[];
  _Float16* xq   = (_Float16*)smem;            // 64x192 f16
  _Float16* xkv  = xq   + CNP * CC;            // 64x192 f16
  _Float16* q16  = xkv  + CNP * CC;            // 64x192 f16 (pre-scaled)
  _Float16* k16  = q16  + CNP * CC;            // 64x192 f16
  _Float16* vT   = k16  + CNP * CC;            // 192x64 f16 (V transposed)
  _Float16* o16  = vT   + CC * CNP;            // 64x192 f16
  float*    S    = (float*)(o16 + CNP * CC);   // 64x64 f32 logits
  _Float16* P    = (_Float16*)(S + CNP * CNP); // 64x64 f16 probs

  int win   = blockIdx.x;           // 0..4095
  int bimg  = win >> 6;
  int w_idx = win & 63;             // window index within image (mask index)
  int tid   = threadIdx.x;
  int wave  = tid >> 5;
  int lane  = tid & 31;
  int coln  = lane & 15;            // D col within tile
  int rowb  = (lane >> 4) * 8;      // D row base within tile

  // --- load normalized window activations (zero-pad rows 49..63) ---
  const _Float16* qw  = qn  + (size_t)win * CNT * CC;
  const _Float16* kvw = kvn + (size_t)win * CNT * CC;
  for (int i = tid; i < CNP * CC; i += CTHR) {
    bool ok = i < CNT * CC;
    xq[i]  = ok ? qw[i]  : (_Float16)0.0f;
    xkv[i] = ok ? kvw[i] : (_Float16)0.0f;
  }
  __syncthreads();

  // --- Q/K/V projections: 3 GEMMs of 64x192x192, 144 WMMA tile jobs ---
  for (int job = wave; job < 144; job += 8) {
    int p  = job / 48;
    int t  = job % 48;
    int mt = t / 12, nt = t % 12;
    const _Float16* Ain; const v16h* WF; const float* bvec;
    if (p == 0)      { Ain = xq;  WF = WqF; bvec = bq; }
    else if (p == 1) { Ain = xkv; WF = WkF; bvec = bk; }
    else             { Ain = xkv; WF = WvF; bvec = bv; }
    v8f acc = {};
#pragma unroll
    for (int kt = 0; kt < 6; ++kt) {
      if (kt < 5) __builtin_prefetch((const void*)(WF + ((kt + 1) * 12 + nt) * 32 + lane), 0, 1);
      v16h a  = load_a_f16(Ain + mt * 16 * CC + kt * 32, CC);
      v16h bb = WF[(kt * 12 + nt) * 32 + lane];
      acc = __builtin_amdgcn_wmma_f32_16x16x32_f16(false, a, false, bb, (short)0, acc, false, false);
    }
    float bias = bvec[nt * 16 + coln];
    if (p == 0) {
      _Float16* dst = q16 + (mt * 16 + rowb) * CC + nt * 16 + coln;
#pragma unroll
      for (int v = 0; v < 8; ++v) dst[v * CC] = (_Float16)((acc[v] + bias) * 0.17677669529663689f);
    } else if (p == 1) {
      _Float16* dst = k16 + (mt * 16 + rowb) * CC + nt * 16 + coln;
#pragma unroll
      for (int v = 0; v < 8; ++v) dst[v * CC] = (_Float16)(acc[v] + bias);
    } else {
      // store V transposed: vT[d][token], 8 contiguous halves per lane
      _Float16* dst = vT + (nt * 16 + coln) * CNP + mt * 16 + rowb;
#pragma unroll
      for (int v = 0; v < 8; ++v) dst[v] = (_Float16)(acc[v] + bias);
    }
  }
  __syncthreads();

  // --- per-head attention ---
  for (int head = 0; head < CNH; ++head) {
    // S = (Q*scale) @ K^T : 4x4 tiles, K-dim = 32 (one WMMA each)
    for (int job = wave; job < 16; job += 8) {
      int mt = job >> 2, ntp = job & 3;
      v8f acc = {};
      v16h a  = load_a_f16(q16 + mt * 16 * CC + head * CHD, CC);
      v16h bb = load_bT_f16(k16 + ntp * 16 * CC + head * CHD, CC);
      acc = __builtin_amdgcn_wmma_f32_16x16x32_f16(false, a, false, bb, (short)0, acc, false, false);
      float* dst = S + (mt * 16 + rowb) * CNP + ntp * 16 + coln;
#pragma unroll
      for (int v = 0; v < 8; ++v) dst[v * CNP] = acc[v];
    }
    __syncthreads();

    // softmax rows with rel-pos bias + shift mask (valid keys m<49 only)
    if (tid < CNP) {
      int r = tid;
      if (r >= CNT) {
        for (int m = 0; m < CNP; ++m) P[r * CNP + m] = (_Float16)0.0f;
      } else {
        const int*   ri   = rel_idx + r * CNT;
        const float* mrow = attn_mask + ((size_t)w_idx * CNT + r) * CNT;
        float mx = -3.0e38f;
        for (int m = 0; m < CNT; ++m) {
          float s = S[r * CNP + m] + rpb[ri[m] * CNH + head] + mrow[m];
          S[r * CNP + m] = s;
          mx = fmaxf(mx, s);
        }
        float sum = 0.0f;
        for (int m = 0; m < CNT; ++m) {
          float e = __expf(S[r * CNP + m] - mx);
          S[r * CNP + m] = e;
          sum += e;
        }
        float inv = 1.0f / sum;
        for (int m = 0; m < CNT; ++m) P[r * CNP + m] = (_Float16)(S[r * CNP + m] * inv);
        for (int m = CNT; m < CNP; ++m) P[r * CNP + m] = (_Float16)0.0f;
      }
    }
    __syncthreads();

    // O = P @ V : 4 M-tiles x 2 D-tiles = 8 jobs, one per wave, K=64 (2 steps)
    {
      int mt = wave >> 1, dt = wave & 1;
      v8f acc = {};
#pragma unroll
      for (int kt = 0; kt < 2; ++kt) {
        v16h a  = load_a_f16(P + mt * 16 * CNP + kt * 32, CNP);
        // B[k=token][n=d] from transposed V: contiguous 16 halves per lane
        v16h bb = load_bT_f16(vT + (head * CHD + dt * 16) * CNP + kt * 32, CNP);
        acc = __builtin_amdgcn_wmma_f32_16x16x32_f16(false, a, false, bb, (short)0, acc, false, false);
      }
      _Float16* dst = o16 + (mt * 16 + rowb) * CC + head * CHD + dt * 16 + coln;
#pragma unroll
      for (int v = 0; v < 8; ++v) dst[v * CC] = (_Float16)acc[v];
    }
    __syncthreads();
  }

  // --- output projection + residual, reverse shift folded into store -------
  int ph0 = (w_idx >> 3) * CWS;
  int pw0 = (w_idx & 7) * CWS;
  for (int job = wave; job < 48; job += 8) {
    int mt = job / 12, nt = job % 12;
    v8f acc = {};
#pragma unroll
    for (int kt = 0; kt < 6; ++kt) {
      if (kt < 5) __builtin_prefetch((const void*)(WprojF + ((kt + 1) * 12 + nt) * 32 + lane), 0, 1);
      v16h a  = load_a_f16(o16 + mt * 16 * CC + kt * 32, CC);
      v16h bb = WprojF[(kt * 12 + nt) * 32 + lane];
      acc = __builtin_amdgcn_wmma_f32_16x16x32_f16(false, a, false, bb, (short)0, acc, false, false);
    }
    int c = nt * 16 + coln;
    float bias = bproj[c];
#pragma unroll
    for (int v = 0; v < 8; ++v) {
      int r = mt * 16 + rowb + v;
      if (r < CNT) {
        int ps = ph0 + r / CWS, qs = pw0 + r % CWS;
        int ho = ps + CSH; if (ho >= CH)  ho -= CH;
        int wo = qs + CSH; if (wo >= CWD) wo -= CWD;
        float resid = query[(((size_t)bimg * CC + c) * CH + ho) * CWD + wo];
        x1[((size_t)bimg * CHW + ho * CWD + wo) * CC + c] = resid + acc[v] + bias;
      }
    }
  }
}

// ------------------------- Kernel 3: fused FFN + residual + NCHW store -----
__global__ __launch_bounds__(CTHR) void swin_ffn_kernel(
    const float* __restrict__ x1,
    const float* __restrict__ gf, const float* __restrict__ betaf,
    const v16h* __restrict__ W1F, const float* __restrict__ b1,
    const v16h* __restrict__ W2F, const float* __restrict__ b2,
    float* __restrict__ out)
{
  extern __shared__ char smem[];
  float*    xf   = (float*)smem;                 // 64x192 f32 (residual)
  _Float16* xn   = (_Float16*)(xf + CNP * CC);   // 64x192 f16 (normed)
  _Float16* h16  = xn + CNP * CC;                // 64x384 f16 (hidden)
  float*    mean_s = (float*)(h16 + CNP * CMLP);
  float*    rstd_s = mean_s + CNP;

  int tid  = threadIdx.x;
  int wave = tid >> 5;
  int lane = tid & 31;
  int coln = lane & 15;
  int rowb = (lane >> 4) * 8;
  size_t t0 = (size_t)blockIdx.x * CNP;

  for (int i = tid; i < CNP * CC; i += CTHR) xf[i] = x1[t0 * CC + i];
  __syncthreads();

  if (tid < CNP) {
    float s = 0.f;
    for (int c = 0; c < CC; ++c) s += xf[tid * CC + c];
    float m = s * (1.0f / CC);
    float v = 0.f;
    for (int c = 0; c < CC; ++c) { float d = xf[tid * CC + c] - m; v += d * d; }
    mean_s[tid] = m;
    rstd_s[tid] = rsqrtf(v * (1.0f / CC) + 1e-5f);
  }
  __syncthreads();

  for (int i = tid; i < CNP * CC; i += CTHR) {
    int r = i / CC, c = i % CC;
    xn[i] = (_Float16)((xf[i] - mean_s[r]) * rstd_s[r] * gf[c] + betaf[c]);
  }
  __syncthreads();

  // GEMM1: (64x192) @ (192x384) + bias, exact GELU
  for (int job = wave; job < 96; job += 8) {
    int mt = job / 24, nt = job % 24;
    v8f acc = {};
#pragma unroll
    for (int kt = 0; kt < 6; ++kt) {
      if (kt < 5) __builtin_prefetch((const void*)(W1F + ((kt + 1) * 24 + nt) * 32 + lane), 0, 1);
      v16h a  = load_a_f16(xn + mt * 16 * CC + kt * 32, CC);
      v16h bb = W1F[(kt * 24 + nt) * 32 + lane];
      acc = __builtin_amdgcn_wmma_f32_16x16x32_f16(false, a, false, bb, (short)0, acc, false, false);
    }
    float bias = b1[nt * 16 + coln];
    _Float16* dst = h16 + (mt * 16 + rowb) * CMLP + nt * 16 + coln;
#pragma unroll
    for (int v = 0; v < 8; ++v) {
      float x = acc[v] + bias;
      float g = 0.5f * x * (1.0f + erff(x * 0.70710678118654752f));
      dst[v * CMLP] = (_Float16)g;
    }
  }
  __syncthreads();

  // GEMM2: (64x384) @ (384x192) + bias + residual, store transposed NCHW
  for (int job = wave; job < 48; job += 8) {
    int mt = job / 12, nt = job % 12;
    v8f acc = {};
#pragma unroll
    for (int kt = 0; kt < 12; ++kt) {
      if (kt < 11) __builtin_prefetch((const void*)(W2F + ((kt + 1) * 12 + nt) * 32 + lane), 0, 1);
      v16h a  = load_a_f16(h16 + mt * 16 * CMLP + kt * 32, CMLP);
      v16h bb = W2F[(kt * 12 + nt) * 32 + lane];
      acc = __builtin_amdgcn_wmma_f32_16x16x32_f16(false, a, false, bb, (short)0, acc, false, false);
    }
    int c = nt * 16 + coln;
    float bias = b2[c];
#pragma unroll
    for (int v = 0; v < 8; ++v) {
      int r = mt * 16 + rowb + v;
      size_t t  = t0 + r;
      float val = acc[v] + bias + xf[r * CC + c];
      size_t bi = t / CHW;
      size_t hw = t % CHW;
      out[(bi * CC + c) * CHW + hw] = val;
    }
  }
}

// --------------------------------- launcher --------------------------------
extern "C" void kernel_launch(void* const* d_in, const int* in_sizes, int n_in,
                              void* d_out, int out_size, void* d_ws, size_t ws_size,
                              hipStream_t stream) {
  (void)in_sizes; (void)n_in; (void)out_size; (void)ws_size;

  const float* query  = (const float*)d_in[0];
  const float* nqg    = (const float*)d_in[1];
  const float* nqb    = (const float*)d_in[2];
  const float* nkvg   = (const float*)d_in[3];
  const float* nkvb   = (const float*)d_in[4];
  const float* Wq     = (const float*)d_in[5];
  const float* bq     = (const float*)d_in[6];
  const float* Wk     = (const float*)d_in[7];
  const float* bk     = (const float*)d_in[8];
  const float* Wv     = (const float*)d_in[9];
  const float* bv     = (const float*)d_in[10];
  const float* rpb    = (const float*)d_in[11];
  const float* Wproj  = (const float*)d_in[12];
  const float* bproj  = (const float*)d_in[13];
  const float* nfg    = (const float*)d_in[14];
  const float* nfb    = (const float*)d_in[15];
  const float* W1     = (const float*)d_in[16];
  const float* b1     = (const float*)d_in[17];
  const float* W2     = (const float*)d_in[18];
  const float* b2     = (const float*)d_in[19];
  const int*   relidx = (const int*)d_in[20];
  const float* amask  = (const float*)d_in[21];
  float* out = (float*)d_out;

  char* ws = (char*)d_ws;
  const size_t X1_BYTES = (size_t)CB * CHW * CC * sizeof(float);      // 154 MB
  const size_t QN_BYTES = (size_t)CBW * CNT * CC * sizeof(_Float16);  // 73.5 MB
  const size_t WQ_HALVES = (size_t)CC * CC;                           // 36864
  const size_t W1_HALVES = (size_t)CC * CMLP;                         // 73728
  float*    x1  = (float*)ws;
  _Float16* qn  = (_Float16*)(ws + X1_BYTES);
  _Float16* kvn = (_Float16*)(ws + X1_BYTES + QN_BYTES);
  _Float16* wsw = (_Float16*)(ws + X1_BYTES + 2 * QN_BYTES);
  _Float16* WqS    = wsw;
  _Float16* WkS    = WqS    + WQ_HALVES;
  _Float16* WvS    = WkS    + WQ_HALVES;
  _Float16* WprojS = WvS    + WQ_HALVES;
  _Float16* W1S    = WprojS + WQ_HALVES;
  _Float16* W2S    = W1S    + W1_HALVES;

  // weight swizzle: f32 row-major -> fragment-ready f16
  swizzle_w_kernel<<<(CC * CC + CTHR - 1) / CTHR, CTHR, 0, stream>>>(Wq, WqS, CC, CC);
  swizzle_w_kernel<<<(CC * CC + CTHR - 1) / CTHR, CTHR, 0, stream>>>(Wk, WkS, CC, CC);
  swizzle_w_kernel<<<(CC * CC + CTHR - 1) / CTHR, CTHR, 0, stream>>>(Wv, WvS, CC, CC);
  swizzle_w_kernel<<<(CC * CC + CTHR - 1) / CTHR, CTHR, 0, stream>>>(Wproj, WprojS, CC, CC);
  swizzle_w_kernel<<<(CC * CMLP + CTHR - 1) / CTHR, CTHR, 0, stream>>>(W1, W1S, CC, CMLP);
  swizzle_w_kernel<<<(CMLP * CC + CTHR - 1) / CTHR, CTHR, 0, stream>>>(W2, W2S, CMLP, CC);

  swin_ln_kernel<<<CB * CH, CTHR, 0, stream>>>(query, nqg, nqb, nkvg, nkvb, qn, kvn);

  size_t attn_smem = (size_t)6 * CNP * CC * sizeof(_Float16)
                   + (size_t)CNP * CNP * sizeof(float)
                   + (size_t)CNP * CNP * sizeof(_Float16);            // 172032 B
  swin_attn_kernel<<<CBW, CTHR, attn_smem, stream>>>(
      qn, kvn, (const v16h*)WqS, bq, (const v16h*)WkS, bk, (const v16h*)WvS, bv,
      rpb, relidx, amask, (const v16h*)WprojS, bproj, query, x1);

  size_t ffn_smem = (size_t)CNP * CC * sizeof(float)
                  + (size_t)CNP * CC * sizeof(_Float16)
                  + (size_t)CNP * CMLP * sizeof(_Float16)
                  + (size_t)CNP * 2 * sizeof(float);                  // 123392 B
  swin_ffn_kernel<<<(CB * CHW) / CNP, CTHR, ffn_smem, stream>>>(
      x1, nfg, nfb, (const v16h*)W1S, b1, (const v16h*)W2S, b2, out);
}
